// se_conv_BiLSTM_20925080666591
// MI455X (gfx1250) — matrix operand order repaired and checked by reading, not compiled, not run on hardware
//
#include <hip/hip_runtime.h>

// ---------------------------------------------------------------------------
// Bidirectional per-pixel (1x1 Conv) LSTM on MI455X (gfx1250, wave32, WMMA).
//   B=128, T=10, P=40, C=300, F=256, 4F=1024 gate cols, rows/step = 5120
// Strategy:
//   * pre-pack weights f32->f16 in exact WMMA B-fragment order (one 32B
//     contiguous load per lane per 16x16 B tile)
//   * pre-convert x to f16 (K padded 300->320); keep h state in f16, c in f32
//   * step kernel: block = 64 rows x 1024 cols, 512 threads = 16 waves,
//     A tiles staged to LDS with global_load_async_to_lds_b128 (ASYNCcnt),
//     GEMM via v_wmma_f32_16x16x32_f16, fused gate epilogue
//   * 20 step launches (fwd stores out[s], bwd adds) = grid-wide recurrence
// ---------------------------------------------------------------------------

typedef _Float16 h8   __attribute__((ext_vector_type(8)));
typedef _Float16 v16h __attribute__((ext_vector_type(16)));
typedef float    v8f  __attribute__((ext_vector_type(8)));

#define Cdim 300
#define KXP  320     // C padded to multiple of 32
#define NCX  10      // KXP/32 k-chunks (input GEMM)
#define Fdim 256
#define NCH  8       // Fdim/32 k-chunks (recurrent GEMM)
#define Tn   10
#define Pn   40
#define NROWS 5120   // B*P
#define NG   1024    // 4*F
#define MROWS 64     // rows per block
#define NTHREADS 512 // 16 waves

// ---------------------------------------------------------------------------
// Pack W (Ksrc x 1024 f32) into f16 B-fragment tiles:
//   dst[((kc*64 + tcol)*32 + lane)*16 + idx] =
//       W[kc*32 + 8*(lane>>4) + (idx&7) + 16*(idx>>3)][tcol*16 + (lane&15)]
// ---------------------------------------------------------------------------
__global__ void pack_w_kernel(const float* __restrict__ W,
                              _Float16* __restrict__ dst,
                              int Ksrc, int nChunks) {
  int e = blockIdx.x * blockDim.x + threadIdx.x;
  int total = nChunks * 64 * 512;
  if (e >= total) return;
  int idx16 = e & 15;
  int lane  = (e >> 4) & 31;
  int tile  = e >> 9;
  int kc    = tile >> 6;
  int tcol  = tile & 63;
  int k   = kc * 32 + ((lane >> 4) << 3) + (idx16 & 7) + ((idx16 >> 3) << 4);
  int col = (tcol << 4) + (lane & 15);
  float v = (k < Ksrc) ? W[(size_t)k * NG + col] : 0.0f;
  dst[e] = (_Float16)v;
}

// x (51200 rows x 300 f32) -> xh (rows x 320 f16, zero padded)
__global__ void pack_x_kernel(const float* __restrict__ x,
                              _Float16* __restrict__ xh, int total) {
  int e = blockIdx.x * blockDim.x + threadIdx.x;
  if (e >= total) return;
  int row = e / KXP, k = e - row * KXP;
  xh[e] = (_Float16)((k < Cdim) ? x[(size_t)row * Cdim + k] : 0.0f);
}

__global__ void zero_u32_kernel(unsigned* __restrict__ p, int n) {
  int i = blockIdx.x * blockDim.x + threadIdx.x;
  if (i < n) p[i] = 0u;
}

// A-fragment (16x32 f16) from a row-major LDS row (two b128 DS loads):
//   idx 0..7  -> K = 8*hi + idx ; idx 8..15 -> K = 16 + 8*hi + (idx-8)
__device__ __forceinline__ v16h frag_a(const _Float16* p, int hi) {
  h8 lo = *(const h8*)(p + 8 * hi);
  h8 hh = *(const h8*)(p + 16 + 8 * hi);
  v16h a;
#pragma unroll
  for (int i = 0; i < 8; ++i) { a[i] = lo[i]; a[8 + i] = hh[i]; }
  return a;
}

template <int ACT>
__device__ __forceinline__ float gate_act(float z) {
  if constexpr (ACT == 0) {  // keras hard_sigmoid
    return fminf(fmaxf(__fmaf_rn(0.2f, z, 0.5f), 0.0f), 1.0f);
  } else {                   // tanh
    return tanhf(z);
  }
}

// ---------------------------------------------------------------------------
// One LSTM time step, one direction. 64 rows x 1024 gate cols per block.
// Wave w (0..15) owns F-columns [16w,16w+16) for all 4 gates, 4 row-tiles.
// ---------------------------------------------------------------------------
template <int ACT, bool ADD>
__global__ __launch_bounds__(NTHREADS)
void lstm_step_kernel(const _Float16* __restrict__ xh,   // (B,T,P,KXP) f16
                      const _Float16* __restrict__ Wkp,  // packed input W
                      const _Float16* __restrict__ Wrp,  // packed recurrent W
                      const float* __restrict__ bias,    // (4F,)
                      _Float16* __restrict__ h16,        // (NROWS,F) f16 state
                      float* __restrict__ cst,           // (NROWS,F) f32 state
                      float* __restrict__ out,           // (B,T,P,F) f32
                      int t_x, int t_out) {
  __shared__ _Float16 Alds[MROWS * KXP + MROWS * Fdim];

  const int tid   = threadIdx.x;
  const int rbase = blockIdx.x * MROWS;
  const unsigned ldsBase = (unsigned)(uintptr_t)(&Alds[0]);

  // ---- async-stage x tile: 64 rows x 640B, b128 granules (ASYNCcnt) ----
#pragma unroll 1
  for (int c = tid; c < MROWS * (KXP / 8); c += NTHREADS) {
    int row = c / (KXP / 8);
    int off = c - row * (KXP / 8);
    int grow = rbase + row;
    int bi = grow / Pn, p = grow - bi * Pn;
    const _Float16* gptr =
        xh + ((size_t)(bi * Tn + t_x) * Pn + p) * KXP + (size_t)off * 8;
    unsigned laddr = ldsBase + (unsigned)(row * (KXP * 2) + off * 16);
    asm volatile("global_load_async_to_lds_b128 %0, %1, off"
                 :: "v"(laddr), "v"(gptr) : "memory");
  }
  // ---- async-stage h tile: contiguous 32KB ----
#pragma unroll 1
  for (int c = tid; c < MROWS * (Fdim / 8); c += NTHREADS) {
    const _Float16* gptr = h16 + (size_t)rbase * Fdim + (size_t)c * 8;
    unsigned laddr = ldsBase + (unsigned)(MROWS * KXP * 2 + c * 16);
    asm volatile("global_load_async_to_lds_b128 %0, %1, off"
                 :: "v"(laddr), "v"(gptr) : "memory");
  }
  asm volatile("s_wait_asynccnt 0x0" ::: "memory");
  __syncthreads();

  const int lane = tid & 31;
  const int hi   = lane >> 4;
  const int nlo  = lane & 15;
  const int w    = tid >> 5;  // n-tile id, 0..15

  v8f acc[4][4];  // [row-tile][gate]
#pragma unroll
  for (int a1 = 0; a1 < 4; ++a1)
#pragma unroll
    for (int a2 = 0; a2 < 4; ++a2)
      acc[a1][a2] = v8f{0.f, 0.f, 0.f, 0.f, 0.f, 0.f, 0.f, 0.f};

  // ---- input contribution: z += x_t @ Wk  (K = 320 padded) ----
  for (int kc = 0; kc < NCX; ++kc) {
    v16h a[4];
#pragma unroll
    for (int rt = 0; rt < 4; ++rt)
      a[rt] = frag_a(&Alds[(size_t)(rt * 16 + nlo) * KXP + kc * 32], hi);
#pragma unroll
    for (int g = 0; g < 4; ++g) {
      const int tcol = g * 16 + w;
      v16h b = *(const v16h*)(Wkp + (((size_t)(kc * 64 + tcol)) << 9) +
                              ((size_t)lane << 4));
#pragma unroll
      for (int rt = 0; rt < 4; ++rt)
        acc[rt][g] = __builtin_amdgcn_wmma_f32_16x16x32_f16(
            false, a[rt], false, b, (short)0, acc[rt][g], false, false);
    }
  }

  // ---- recurrent contribution: z += h @ Wr  (K = 256) ----
  const _Float16* Hlds = &Alds[MROWS * KXP];
  for (int kc = 0; kc < NCH; ++kc) {
    v16h a[4];
#pragma unroll
    for (int rt = 0; rt < 4; ++rt)
      a[rt] = frag_a(&Hlds[(size_t)(rt * 16 + nlo) * Fdim + kc * 32], hi);
#pragma unroll
    for (int g = 0; g < 4; ++g) {
      const int tcol = g * 16 + w;
      v16h b = *(const v16h*)(Wrp + (((size_t)(kc * 64 + tcol)) << 9) +
                              ((size_t)lane << 4));
#pragma unroll
      for (int rt = 0; rt < 4; ++rt)
        acc[rt][g] = __builtin_amdgcn_wmma_f32_16x16x32_f16(
            false, a[rt], false, b, (short)0, acc[rt][g], false, false);
    }
  }

  // ---- epilogue: bias, gates (keras i,f,c,o), state update, output ----
  // C/D layout: VGPR r -> M = r + 8*hi, N = lane&15 within the 16x16 tile.
  const int ncol = w * 16 + nlo;  // 0..255 within F
  const float bi_ = bias[ncol];
  const float bf_ = bias[256 + ncol];
  const float bg_ = bias[512 + ncol];
  const float bo_ = bias[768 + ncol];
#pragma unroll
  for (int rt = 0; rt < 4; ++rt) {
#pragma unroll
    for (int r = 0; r < 8; ++r) {
      const int grow = rbase + rt * 16 + hi * 8 + r;
      const size_t si = (size_t)grow * Fdim + ncol;
      float zi = acc[rt][0][r] + bi_;
      float zf = acc[rt][1][r] + bf_;
      float zg = acc[rt][2][r] + bg_;
      float zo = acc[rt][3][r] + bo_;
      float iv = gate_act<ACT>(zi);
      float fv = gate_act<ACT>(zf);
      float gv = tanhf(zg);
      float ov = gate_act<ACT>(zo);
      float cn = fv * cst[si] + iv * gv;
      float hn = ov * tanhf(cn);
      cst[si] = cn;
      h16[si] = (_Float16)hn;
      int bidx = grow / Pn, p = grow - bidx * Pn;
      size_t oi = (((size_t)bidx * Tn + t_out) * Pn + p) * (size_t)Fdim + ncol;
      if constexpr (ADD) out[oi] += hn; else out[oi] = hn;
    }
  }
}

extern "C" void kernel_launch(void* const* d_in, const int* in_sizes, int n_in,
                              void* d_out, int out_size, void* d_ws, size_t ws_size,
                              hipStream_t stream) {
  (void)in_sizes; (void)n_in; (void)out_size; (void)ws_size;
  const float* x   = (const float*)d_in[0];
  const float* WkF = (const float*)d_in[1];
  const float* WrF = (const float*)d_in[2];
  const float* bF  = (const float*)d_in[3];
  const float* WkB = (const float*)d_in[4];
  const float* WrB = (const float*)d_in[5];
  const float* bB  = (const float*)d_in[6];
  float* out = (float*)d_out;

  // ---- workspace layout ----
  const size_t SZ_WK = (size_t)NCX * 64 * 512;  // packed input-W halves
  const size_t SZ_WR = (size_t)NCH * 64 * 512;  // packed recurrent-W halves
  const size_t SZ_XH = (size_t)NROWS * Tn * KXP; // padded f16 x halves
  const size_t SSZ   = (size_t)NROWS * Fdim;

  _Float16* wkf = (_Float16*)d_ws;
  _Float16* wrf = wkf + SZ_WK;
  _Float16* wkb = wrf + SZ_WR;
  _Float16* wrb = wkb + SZ_WK;
  _Float16* xhp = wrb + SZ_WR;        // 16,384,000 halves
  _Float16* hF  = xhp + SZ_XH;        // f16 states (zero-init region start)
  _Float16* hB  = hF + SSZ;
  float*    cF  = (float*)(hB + SSZ); // f32 states
  float*    cB  = cF + SSZ;

  // ---- pre-pack weights + x, zero states ----
  {
    int nk = (int)SZ_WK, nr = (int)SZ_WR;
    pack_w_kernel<<<(nk + 255) / 256, 256, 0, stream>>>(WkF, wkf, Cdim, NCX);
    pack_w_kernel<<<(nr + 255) / 256, 256, 0, stream>>>(WrF, wrf, Fdim, NCH);
    pack_w_kernel<<<(nk + 255) / 256, 256, 0, stream>>>(WkB, wkb, Cdim, NCX);
    pack_w_kernel<<<(nr + 255) / 256, 256, 0, stream>>>(WrB, wrb, Fdim, NCH);
    int nx = (int)SZ_XH;
    pack_x_kernel<<<(nx + 255) / 256, 256, 0, stream>>>(x, xhp, nx);
    // zero hF, hB (f16) + cF, cB (f32) in one pass: bytes = 2*SSZ*2 + 2*SSZ*4
    int nz = (int)((2 * SSZ * 2 + 2 * SSZ * 4) / 4);
    zero_u32_kernel<<<(nz + 255) / 256, 256, 0, stream>>>((unsigned*)hF, nz);
  }

  // ---- recurrence: fwd consumes x[s], bwd x[T-1-s]; both land on out[s] ----
  const int nblk = NROWS / MROWS;  // 80
  for (int s = 0; s < Tn; ++s) {
    lstm_step_kernel<0, false><<<nblk, NTHREADS, 0, stream>>>(
        xhp, wkf, wrf, bF, hF, cF, out, s, s);
    lstm_step_kernel<1, true><<<nblk, NTHREADS, 0, stream>>>(
        xhp, wkb, wrb, bB, hB, cB, out, Tn - 1 - s, s);
  }
}
